// CosHead_48799418417617
// MI455X (gfx1250) — matrix operand order repaired and checked by reading, not compiled, not run on hardware
//
#include <hip/hip_runtime.h>

// Problem constants (fixed by the reference setup):
// x: [8, 256, 128, 128] f32 ; weights: [80, 256] f32 ; asf: [80] f32
// out: [8, 80, 128, 128] f32
#define B_    8
#define D_    256
#define H_    128
#define W_    128
#define C_    80
#define HW_   (H_ * W_)          // 16384
#define NPOS  (B_ * HW_)         // 131072 positions
#define WROW  264                // LDS row stride in bf16 elems (528 B -> bank step 4, no conflicts)

typedef __attribute__((ext_vector_type(16))) __bf16 v16bf;
typedef __attribute__((ext_vector_type(8)))  float  v8f;

// round-to-nearest-even f32 -> bf16 bits (used only in the tiny prep kernel)
__device__ __forceinline__ unsigned short f32_to_bf16_bits(float f) {
    unsigned int u = __float_as_uint(f);
    return (unsigned short)((u + 0x7fffu + ((u >> 16) & 1u)) >> 16);
}

// ---------------------------------------------------------------------------
// Kernel 1: fold  wf[c,d] = (w[c,d]/max(||w_c||,eps)) * asf[c] * 10  -> bf16
// One block per channel c; 256 threads = one thread per d.
// ---------------------------------------------------------------------------
__global__ __launch_bounds__(256) void fold_weights(
    const float* __restrict__ w, const float* __restrict__ asf,
    unsigned short* __restrict__ wf)
{
    const int c = blockIdx.x;
    const int t = threadIdx.x;              // == d
    float v = w[c * D_ + t];
    float s = v * v;
#pragma unroll
    for (int o = 16; o > 0; o >>= 1) s += __shfl_xor(s, o, 32);

    __shared__ float part[8];
    __shared__ float tot;
    if ((t & 31) == 0) part[t >> 5] = s;
    __syncthreads();
    if (t == 0) {
        float q = 0.f;
#pragma unroll
        for (int i = 0; i < 8; ++i) q += part[i];
        tot = q;
    }
    __syncthreads();

    float inv = 1.0f / fmaxf(sqrtf(tot), 1e-8f);
    wf[c * D_ + t] = f32_to_bf16_bits(v * inv * asf[c] * 10.0f);
}

// ---------------------------------------------------------------------------
// Kernel 2: fused normalized GEMM.
// Block = 256 threads = 8 waves. Wave owns 16 consecutive positions (same b,h;
// w0..w0+15) x all 80 channels (5 WMMA N-tiles). K = 256 in 8 steps of 32.
// ---------------------------------------------------------------------------
__global__ __launch_bounds__(256) void coshead_wmma(
    const float* __restrict__ x, const unsigned short* __restrict__ wf,
    float* __restrict__ out)
{
    __shared__ unsigned short wlds[C_ * WROW];   // 42240 B, padded rows
    __shared__ float tbuf[8][16 * 20];           // per-wave 16x16 transpose, 20-float rows

    const int tid  = threadIdx.x;
    const int lane = tid & 31;
    const int wv   = tid >> 5;

    // ---- stage folded weights into LDS (b128 loads/stores) ----
    {
        const uint4* src = (const uint4*)wf;     // 80*256*2 B = 2560 uint4
#pragma unroll
        for (int s = 0; s < 10; ++s) {
            int j = tid + 256 * s;
            uint4 vq = src[j];
            int c = j >> 5;                      // 32 uint4 per 256-elem row
            *(uint4*)&wlds[c * WROW + (j & 31) * 8] = vq;
        }
    }
    __syncthreads();

    // ---- this wave's 16 positions ----
    const int P0  = (blockIdx.x * 8 + wv) * 16;  // multiple of 16, so same (b,h)
    const int b   = P0 >> 14;                    // / HW_
    const int rem = P0 & (HW_ - 1);              // h*W + w0
    const int m     = lane & 15;                 // row within M-tile
    const int khalf = (lane >> 4) * 8;           // A-fragment K half per ISA layout

    const float* pxk = x + (size_t)b * D_ * HW_ + rem + m + (size_t)khalf * HW_;
    const unsigned short* wbase = &wlds[(lane & 15) * WROW + (lane >> 4) * 8];

    const v8f vzero = {0.f, 0.f, 0.f, 0.f, 0.f, 0.f, 0.f, 0.f};
    v8f acc[5];
#pragma unroll
    for (int t = 0; t < 5; ++t) acc[t] = vzero;
    float ssq = 0.f;

    union BU { uint4 q[2]; v16bf v; };

#pragma unroll
    for (int k0 = 0; k0 < D_; k0 += 32) {
        const float* p = pxk + (size_t)k0 * HW_;
        float xv[16];
#pragma unroll
        for (int e = 0; e < 8; ++e) xv[e]     = p[(size_t)e * HW_];         // K = k0+khalf+e
#pragma unroll
        for (int e = 0; e < 8; ++e) xv[8 + e] = p[(size_t)(16 + e) * HW_];  // K = k0+16+khalf+e

        v16bf a;
#pragma unroll
        for (int e = 0; e < 16; ++e) {
            ssq = fmaf(xv[e], xv[e], ssq);
            a[e] = (__bf16)xv[e];                // native RNE cvt (v_cvt_*bf16*)
        }

#pragma unroll
        for (int t = 0; t < 5; ++t) {
            const unsigned short* wp = wbase + t * 16 * WROW + k0;
            BU bf;
            bf.q[0] = *(const uint4*)(wp);
            bf.q[1] = *(const uint4*)(wp + 16);
            acc[t] = __builtin_amdgcn_wmma_f32_16x16x32_bf16(
                false, a, false, bf.v, (short)0, acc[t], false, false);
        }
    }

    // ---- finish ||x|| across the two K-halves, broadcast per output row ----
    float ssqf = ssq + __shfl_xor(ssq, 16, 32);        // lane L holds full ssq of row L%16
    float invn = 1.0f / fmaxf(sqrtf(ssqf), 1e-8f);
    float rn[8];
#pragma unroll
    for (int r = 0; r < 8; ++r)
        rn[r] = __shfl(invn, r + ((lane >> 4) << 3), 32);  // row = r + 8*(lane/16)

    // ---- scale, transpose through LDS, store coalesced float4 along w ----
    float* tb = tbuf[wv];
    const size_t outbase = (size_t)b * C_ * HW_ + rem;
    const int cc = lane >> 2;          // 0..7
    const int mm = (lane & 3) * 4;     // 0,4,8,12
#pragma unroll
    for (int t = 0; t < 5; ++t) {
#pragma unroll
        for (int r = 0; r < 8; ++r)
            tb[(lane & 15) * 20 + ((lane >> 4) << 3) + r] = acc[t][r] * rn[r];
        __syncthreads();
#pragma unroll
        for (int j = 0; j < 2; ++j) {
            int c = t * 16 + j * 8 + cc;
            float4 vv = *(const float4*)&tb[(j * 8 + cc) * 20 + mm];
            *(float4*)&out[outbase + (size_t)c * HW_ + mm] = vv;
        }
        __syncthreads();
    }
}

// ---------------------------------------------------------------------------
extern "C" void kernel_launch(void* const* d_in, const int* in_sizes, int n_in,
                              void* d_out, int out_size, void* d_ws, size_t ws_size,
                              hipStream_t stream) {
    const float* x   = (const float*)d_in[0];
    const float* w   = (const float*)d_in[1];
    const float* asf = (const float*)d_in[2];
    float* out = (float*)d_out;
    unsigned short* wf = (unsigned short*)d_ws;   // 80*256 bf16 = 40960 B

    fold_weights<<<C_, 256, 0, stream>>>(w, asf, wf);
    coshead_wmma<<<NPOS / 128, 256, 0, stream>>>(x, wf, out);
}
